// PerceiverResampler_50216757625236
// MI455X (gfx1250) — matrix operand order repaired and checked by reading, not compiled, not run on hardware
//
#include <hip/hip_runtime.h>
#include <hip/hip_bf16.h>
#include <math.h>

// ---------------- types & helpers ----------------
typedef __bf16 bf16_t;
typedef bf16_t bf16x16 __attribute__((ext_vector_type(16)));
typedef float  f32x8   __attribute__((ext_vector_type(8)));
typedef float  f32x4   __attribute__((ext_vector_type(4)));
typedef unsigned short us8 __attribute__((ext_vector_type(8)));
typedef int    v4i_t   __attribute__((vector_size(16)));

union FragB16 { bf16x16 v; us8 h[2]; };

__device__ __forceinline__ unsigned short f2bf(float f) {
  unsigned u = __float_as_uint(f);
  u += 0x7fffu + ((u >> 16) & 1u);          // round-to-nearest-even
  return (unsigned short)(u >> 16);
}
__device__ __forceinline__ float bf2f(unsigned short h) {
  return __uint_as_float(((unsigned)h) << 16);
}

// ---- CDNA5 async global->LDS (guarded: falls back to sync staging if absent) ----
#if __has_builtin(__builtin_amdgcn_global_load_async_to_lds_b128) && \
    __has_builtin(__builtin_amdgcn_s_wait_asynccnt)
#define USE_ASYNC 1
__device__ __forceinline__ void async_b128(const void* g, void* l) {
  __builtin_amdgcn_global_load_async_to_lds_b128(
      (__attribute__((address_space(1))) v4i_t*)g,
      (__attribute__((address_space(3))) v4i_t*)l, 0, 0);
}
#else
#define USE_ASYNC 0
#endif

#define DIMD   768
#define INNER  512
#define FFH    3072
#define NBATCH 4
#define NZ     64
#define NXTOK  25088          // T*F*v = 8*16*196
#define NFTOK  25152          // NXTOK + NZ
#define NSPLIT 8
#define NTILES (NFTOK / 64)   // 393

// ---------------- prep: x + frame_emb + time_emb, layernorm stats, bf16 x-hat ----------------
__global__ __launch_bounds__(256)
void prep_xhat_kernel(const float* __restrict__ x, const float* __restrict__ fe,
                      const float* __restrict__ te, unsigned short* __restrict__ xhat)
{
  const int T = 8, F = 16, V = 196;
  int r   = blockIdx.x;                       // token row, 0..B*NXTOK-1
  int rem = r % (T * F * V);
  int t   = rem / (F * V);
  int f   = (rem % (F * V)) / V;
  const float* xp = x + (size_t)r * DIMD;
  __shared__ float red[256];
  float vals[3];
  float s = 0.f;
  #pragma unroll
  for (int i = 0; i < 3; i++) {
    int d = threadIdx.x + i * 256;
    float vv = xp[d] + fe[f * DIMD + d] + te[t * DIMD + d];
    vals[i] = vv; s += vv;
  }
  red[threadIdx.x] = s; __syncthreads();
  for (int o = 128; o > 0; o >>= 1) { if (threadIdx.x < o) red[threadIdx.x] += red[threadIdx.x + o]; __syncthreads(); }
  float mean = red[0] * (1.f / 768.f);
  __syncthreads();
  float s2 = 0.f;
  #pragma unroll
  for (int i = 0; i < 3; i++) { float dd = vals[i] - mean; s2 += dd * dd; }
  red[threadIdx.x] = s2; __syncthreads();
  for (int o = 128; o > 0; o >>= 1) { if (threadIdx.x < o) red[threadIdx.x] += red[threadIdx.x + o]; __syncthreads(); }
  float rstd = rsqrtf(red[0] * (1.f / 768.f) + 1e-5f);
  #pragma unroll
  for (int i = 0; i < 3; i++) {
    int d = threadIdx.x + i * 256;
    xhat[(size_t)r * DIMD + d] = f2bf((vals[i] - mean) * rstd);
  }
}

// ---------------- generic row layernorm (256 rows of lat), bf16 or f32 out ----------------
template <bool OUTBF16>
__global__ __launch_bounds__(256)
void layernorm_rows_kernel(const float* __restrict__ in, const float* __restrict__ g,
                           const float* __restrict__ b, void* __restrict__ out)
{
  int r = blockIdx.x;
  const float* xp = in + (size_t)r * DIMD;
  __shared__ float red[256];
  float vals[3];
  float s = 0.f;
  #pragma unroll
  for (int i = 0; i < 3; i++) { float vv = xp[threadIdx.x + i * 256]; vals[i] = vv; s += vv; }
  red[threadIdx.x] = s; __syncthreads();
  for (int o = 128; o > 0; o >>= 1) { if (threadIdx.x < o) red[threadIdx.x] += red[threadIdx.x + o]; __syncthreads(); }
  float mean = red[0] * (1.f / 768.f);
  __syncthreads();
  float s2 = 0.f;
  #pragma unroll
  for (int i = 0; i < 3; i++) { float dd = vals[i] - mean; s2 += dd * dd; }
  red[threadIdx.x] = s2; __syncthreads();
  for (int o = 128; o > 0; o >>= 1) { if (threadIdx.x < o) red[threadIdx.x] += red[threadIdx.x + o]; __syncthreads(); }
  float rstd = rsqrtf(red[0] * (1.f / 768.f) + 1e-5f);
  #pragma unroll
  for (int i = 0; i < 3; i++) {
    int d = threadIdx.x + i * 256;
    float nv = (vals[i] - mean) * rstd * g[d] + b[d];
    if (OUTBF16) ((unsigned short*)out)[(size_t)r * DIMD + d] = f2bf(nv);
    else         ((float*)out)[(size_t)r * DIMD + d] = nv;
  }
}

__global__ void init_lat_kernel(const float* __restrict__ latents, float* __restrict__ lat)
{
  int idx = blockIdx.x * 256 + threadIdx.x;   // B*NZ*D elements
  int r = idx / DIMD, d = idx % DIMD;
  lat[idx] = latents[(r & (NZ - 1)) * DIMD + d];
}

// ---------------- fold media-LN gamma/beta into K/V weights, bf16, transposed [N][K] ----------------
// Wf[n][k] = bf16(g[k]*W[k][n]);  bias[n] = sum_k b[k]*W[k][n]
__global__ __launch_bounds__(256)
void fold_weights_kernel(const float* __restrict__ W, const float* __restrict__ g,
                         const float* __restrict__ b, unsigned short* __restrict__ Wt,
                         float* __restrict__ bias)
{
  int n = blockIdx.x;                 // 0..INNER-1
  __shared__ float red[256];
  float acc = 0.f;
  for (int k = threadIdx.x; k < DIMD; k += 256) {
    float w = W[(size_t)k * INNER + n];
    Wt[(size_t)n * DIMD + k] = f2bf(g[k] * w);
    acc += b[k] * w;
  }
  red[threadIdx.x] = acc; __syncthreads();
  for (int o = 128; o > 0; o >>= 1) { if (threadIdx.x < o) red[threadIdx.x] += red[threadIdx.x + o]; __syncthreads(); }
  if (threadIdx.x == 0) bias[n] = red[0];
}

// shared WMMA micro-step: 16-row A strip x 128 B cols from padded LDS tiles (40-elem rows)
__device__ __forceinline__ void kv_compute_step(const unsigned short* bufA, const unsigned short* bufB,
                                                int wave, int l16, int half, f32x8 (&acc)[8])
{
  FragB16 af;
  const int arow = wave * 16 + l16;
  af.h[0] = *(const us8*)&bufA[arow * 40 + half * 8];        // k = 8h + 0..7
  af.h[1] = *(const us8*)&bufA[arow * 40 + 16 + half * 8];   // k = 16 + 8h + 0..7
  #pragma unroll
  for (int nt = 0; nt < 8; nt++) {
    FragB16 bf;
    const int brow = nt * 16 + l16;
    bf.h[0] = *(const us8*)&bufB[brow * 40 + half * 8];      // k = 8h + 0..7
    bf.h[1] = *(const us8*)&bufB[brow * 40 + 16 + half * 8]; // k = 16 + 8h + 0..7
    acc[nt] = __builtin_amdgcn_wmma_f32_16x16x32_bf16(
        false, af.v, false, bf.v, (short)0, acc[nt], false, false);
  }
}

// ---------------- K/V projection GEMM: A=xhat bf16 [Mtot,768], Bt bf16 [512][768] ----------------
// Double-buffered GLOBAL_LOAD_ASYNC_TO_LDS_B128 staging (overlaps next slice with WMMA).
__global__ __launch_bounds__(256)
void gemm_kv_kernel(const unsigned short* __restrict__ A,
                    const unsigned short* __restrict__ Bt,
                    const float* __restrict__ bias,
                    unsigned short* __restrict__ Out)
{
  const int tid  = threadIdx.x;
  const int wave = tid >> 5, lane = tid & 31;
  const int half = lane >> 4, l16 = lane & 15;
  const int m0 = blockIdx.x * 128;
  const int n0 = blockIdx.y * 128;
  const int row0 = tid >> 2;              // 0..63 (second chunk handles +64)
  const int kc0  = (tid & 3) * 8;
  const int NK = DIMD / 32;               // 24 k-steps

  f32x8 acc[8];
  #pragma unroll
  for (int nt = 0; nt < 8; nt++)
    #pragma unroll
    for (int j = 0; j < 8; j++) acc[nt][j] = 0.f;

#if USE_ASYNC
  __shared__ alignas(16) unsigned short lA[2][128 * 40];
  __shared__ alignas(16) unsigned short lB[2][128 * 40];
  // per-thread: 4 async 16B copies per k-step (A rows row0,row0+64; B rows row0,row0+64)
  #define KV_ISSUE(s, ks)                                                                 \
    do {                                                                                  \
      async_b128(A  + (size_t)(m0 + row0)      * DIMD + (ks) + kc0, &lA[s][ row0       * 40 + kc0]); \
      async_b128(A  + (size_t)(m0 + row0 + 64) * DIMD + (ks) + kc0, &lA[s][(row0 + 64) * 40 + kc0]); \
      async_b128(Bt + (size_t)(n0 + row0)      * DIMD + (ks) + kc0, &lB[s][ row0       * 40 + kc0]); \
      async_b128(Bt + (size_t)(n0 + row0 + 64) * DIMD + (ks) + kc0, &lB[s][(row0 + 64) * 40 + kc0]); \
    } while (0)

  KV_ISSUE(0, 0);
  __builtin_amdgcn_s_wait_asynccnt(0);
  __syncthreads();
  for (int i = 0; i < NK; i++) {
    if (i + 1 < NK) KV_ISSUE((i + 1) & 1, (i + 1) * 32);   // prefetch next slice
    kv_compute_step(lA[i & 1], lB[i & 1], wave, l16, half, acc);
    __builtin_amdgcn_s_wait_asynccnt(0);
    __syncthreads();
  }
  #undef KV_ISSUE
#else
  __shared__ alignas(16) unsigned short sA[128 * 40];
  __shared__ alignas(16) unsigned short sB[128 * 40];
  for (int i = 0; i < NK; i++) {
    int ks = i * 32;
    __syncthreads();
    *(us8*)&sA[ row0       * 40 + kc0] = *(const us8*)(A  + (size_t)(m0 + row0)      * DIMD + ks + kc0);
    *(us8*)&sA[(row0 + 64) * 40 + kc0] = *(const us8*)(A  + (size_t)(m0 + row0 + 64) * DIMD + ks + kc0);
    *(us8*)&sB[ row0       * 40 + kc0] = *(const us8*)(Bt + (size_t)(n0 + row0)      * DIMD + ks + kc0);
    *(us8*)&sB[(row0 + 64) * 40 + kc0] = *(const us8*)(Bt + (size_t)(n0 + row0 + 64) * DIMD + ks + kc0);
    __syncthreads();
    kv_compute_step(sA, sB, wave, l16, half, acc);
  }
#endif

  #pragma unroll
  for (int nt = 0; nt < 8; nt++) {
    float bn = bias[n0 + nt * 16 + l16];
    #pragma unroll
    for (int r = 0; r < 8; r++) {
      int m = m0 + wave * 16 + r + half * 8;
      int n = n0 + nt * 16 + l16;
      size_t orow = (size_t)(m / NXTOK) * NFTOK + (m % NXTOK);   // media rows of K/V buffer
      Out[orow * INNER + n] = f2bf(acc[nt][r] + bn);
    }
  }
}

// ---------------- generic bf16 WMMA GEMM (latent side): C = A[M,K](bf16) @ W[K,N](f32->bf16) ----------------
// OUT_MODE: 0 = bf16 store, 1 = f32 residual add, 2 = bf16 store with exact GELU.
template <int OUT_MODE>
__global__ __launch_bounds__(256)
void gemm_bf16_kernel(const unsigned short* __restrict__ A, int lda,
                      const float* __restrict__ W, int ldw,
                      void* __restrict__ Out, int ldo,
                      int K, int rps, int seg, int roff)
{
  __shared__ alignas(16) unsigned short lA[128 * 40];   // [row][k] pad->80B rows
  __shared__ alignas(16) unsigned short lB[128 * 40];   // transposed [n][k]
  const int tid  = threadIdx.x;
  const int wave = tid >> 5, lane = tid & 31;
  const int half = lane >> 4, l16 = lane & 15;
  const int m0 = blockIdx.x * 128;
  const int n0 = blockIdx.y * 128;

  f32x8 acc[8];
  #pragma unroll
  for (int nt = 0; nt < 8; nt++)
    #pragma unroll
    for (int j = 0; j < 8; j++) acc[nt][j] = 0.f;

  for (int ks = 0; ks < K; ks += 32) {
    __syncthreads();
    #pragma unroll
    for (int i = 0; i < 2; i++) {
      int idx = tid + i * 256;
      int row = idx >> 2, kc = (idx & 3) * 8;
      *(us8*)&lA[row * 40 + kc] = *(const us8*)(A + (size_t)(m0 + row) * lda + ks + kc);
    }
    #pragma unroll
    for (int i = 0; i < 4; i++) {
      int idx = tid + i * 256;
      int kr = idx >> 5, nc = (idx & 31) * 4;
      f32x4 wv = *(const f32x4*)(W + (size_t)(ks + kr) * ldw + n0 + nc);
      #pragma unroll
      for (int j = 0; j < 4; j++) lB[(nc + j) * 40 + kr] = f2bf(wv[j]);
    }
    __syncthreads();
    kv_compute_step(lA, lB, wave, l16, half, acc);
  }

  #pragma unroll
  for (int nt = 0; nt < 8; nt++) {
    #pragma unroll
    for (int r = 0; r < 8; r++) {
      int m = m0 + wave * 16 + r + half * 8;
      int n = n0 + nt * 16 + l16;
      float val = acc[nt][r];
      size_t orow = (size_t)(m / rps) * seg + (m % rps) + roff;
      if (OUT_MODE == 1) {
        ((float*)Out)[orow * ldo + n] += val;
      } else {
        if (OUT_MODE == 2)
          val = 0.5f * val * (1.0f + erff(val * 0.70710678118654752f));
        ((unsigned short*)Out)[orow * ldo + n] = f2bf(val);
      }
    }
  }
}

// ---------------- flash attention, split over f; per block: one (b,h), 4 waves, q rows 0..63 ----------------
__global__ __launch_bounds__(128)
void flash_attn_kernel(const unsigned short* __restrict__ Q,   // [B*64, 512] bf16
                       const unsigned short* __restrict__ Kb,  // [B*NF, 512] bf16
                       const unsigned short* __restrict__ Vb,  // [B*NF, 512] bf16
                       float* __restrict__ Opart, float* __restrict__ Mpart,
                       float* __restrict__ Spart, int tiles_per_split)
{
  const int bh = blockIdx.x;            // 0..31
  const int split = blockIdx.y;         // 0..NSPLIT-1
  const int b = bh >> 3, h = bh & 7;
  const int tid = threadIdx.x, wave = tid >> 5, lane = tid & 31;
  const int half = lane >> 4, l16 = lane & 15;

  __shared__ alignas(16) unsigned short lK[64 * 72];      // [f][d]
  __shared__ alignas(16) unsigned short lV[64 * 72];      // transposed [d][f]
  __shared__ alignas(16) unsigned short lP[4 * 16 * 72];  // per-wave P tile [q][f]
  __shared__ float lM[64], lS[64];

  if (tid < 64) { lM[tid] = -INFINITY; lS[tid] = 0.f; }

  f32x8 O[4];
  #pragma unroll
  for (int nt = 0; nt < 4; nt++)
    #pragma unroll
    for (int j = 0; j < 8; j++) O[nt][j] = 0.f;

  FragB16 qf[2];
  {
    const unsigned short* qp = Q + (size_t)(b * 64 + wave * 16 + l16) * INNER + h * 64;
    #pragma unroll
    for (int c = 0; c < 2; c++) {
      qf[c].h[0] = *(const us8*)(qp + c * 32 + half * 8);
      qf[c].h[1] = *(const us8*)(qp + c * 32 + 16 + half * 8);
    }
  }

  const float SCALE = 0.125f;   // DIM_HEAD^-0.5
  int t0 = split * tiles_per_split;
  int t1 = t0 + tiles_per_split; if (t1 > NTILES) t1 = NTILES;

  for (int t = t0; t < t1; ++t) {
    int f0 = t * 64;
    __syncthreads();
    #pragma unroll
    for (int i = 0; i < 4; i++) {
      int idx = tid + i * 128;
      int fr = idx >> 3, dc = (idx & 7) * 8;
      size_t gofs = ((size_t)b * NFTOK + f0 + fr) * INNER + h * 64 + dc;
      us8 kv = *(const us8*)(Kb + gofs);
      *(us8*)&lK[fr * 72 + dc] = kv;
      us8 vv = *(const us8*)(Vb + gofs);
      #pragma unroll
      for (int j = 0; j < 8; j++) lV[(dc + j) * 72 + fr] = vv[j];
    }
    __syncthreads();

    f32x8 S[4];
    #pragma unroll
    for (int nt = 0; nt < 4; nt++) {
      f32x8 s;
      #pragma unroll
      for (int j = 0; j < 8; j++) s[j] = 0.f;
      #pragma unroll
      for (int c = 0; c < 2; c++) {
        FragB16 kf;
        int krow = nt * 16 + l16;
        kf.h[0] = *(const us8*)&lK[krow * 72 + c * 32 + half * 8];
        kf.h[1] = *(const us8*)&lK[krow * 72 + c * 32 + 16 + half * 8];
        s = __builtin_amdgcn_wmma_f32_16x16x32_bf16(
            false, qf[c].v, false, kf.v, (short)0, s, false, false);
      }
      S[nt] = s;
    }

    #pragma unroll
    for (int r = 0; r < 8; r++) {
      int ro = wave * 16 + r + half * 8;
      float smax = -INFINITY;
      #pragma unroll
      for (int nt = 0; nt < 4; nt++) smax = fmaxf(smax, S[nt][r] * SCALE);
      #pragma unroll
      for (int off = 1; off < 16; off <<= 1) smax = fmaxf(smax, __shfl_xor(smax, off, 16));
      float oldm = lM[ro];
      float newm = fmaxf(oldm, smax);
      float corr = __expf(oldm - newm);
      float rsum = 0.f;
      #pragma unroll
      for (int nt = 0; nt < 4; nt++) {
        float p = __expf(S[nt][r] * SCALE - newm);
        lP[wave * (16 * 72) + (r + half * 8) * 72 + nt * 16 + l16] = f2bf(p);
        rsum += p;
      }
      #pragma unroll
      for (int off = 1; off < 16; off <<= 1) rsum += __shfl_xor(rsum, off, 16);
      if (l16 == 0) { lM[ro] = newm; lS[ro] = lS[ro] * corr + rsum; }
      #pragma unroll
      for (int nt = 0; nt < 4; nt++) O[nt][r] *= corr;
    }

    #pragma unroll
    for (int c = 0; c < 2; c++) {
      FragB16 pf;
      pf.h[0] = *(const us8*)&lP[wave * (16 * 72) + l16 * 72 + c * 32 + half * 8];
      pf.h[1] = *(const us8*)&lP[wave * (16 * 72) + l16 * 72 + c * 32 + 16 + half * 8];
      #pragma unroll
      for (int nt = 0; nt < 4; nt++) {
        FragB16 vf;
        int drow = nt * 16 + l16;
        vf.h[0] = *(const us8*)&lV[drow * 72 + c * 32 + half * 8];
        vf.h[1] = *(const us8*)&lV[drow * 72 + c * 32 + 16 + half * 8];
        O[nt] = __builtin_amdgcn_wmma_f32_16x16x32_bf16(
            false, pf.v, false, vf.v, (short)0, O[nt], false, false);
      }
    }
  }
  __syncthreads();

  size_t base = ((size_t)bh * NSPLIT + split) * (64 * 64);
  #pragma unroll
  for (int nt = 0; nt < 4; nt++)
    #pragma unroll
    for (int r = 0; r < 8; r++) {
      int q = wave * 16 + r + half * 8;
      int d = nt * 16 + l16;
      Opart[base + q * 64 + d] = O[nt][r];
    }
  if (tid < 64) {
    size_t sb = ((size_t)bh * NSPLIT + split) * 64 + tid;
    Mpart[sb] = lM[tid];
    Spart[sb] = lS[tid];
  }
}

// ---------------- combine split partials (log-sum-exp merge) -> attn bf16 [B*64, 512] ----------------
__global__ void attn_combine_kernel(const float* __restrict__ Opart, const float* __restrict__ Mpart,
                                    const float* __restrict__ Spart, unsigned short* __restrict__ attn)
{
  int bh = blockIdx.x; int b = bh >> 3, h = bh & 7;
  int q = threadIdx.x;                       // 0..63
  float M = -INFINITY;
  for (int s = 0; s < NSPLIT; s++) M = fmaxf(M, Mpart[((size_t)bh * NSPLIT + s) * 64 + q]);
  float w[NSPLIT], S = 0.f;
  for (int s = 0; s < NSPLIT; s++) {
    float ws = __expf(Mpart[((size_t)bh * NSPLIT + s) * 64 + q] - M);
    w[s] = ws;
    S += ws * Spart[((size_t)bh * NSPLIT + s) * 64 + q];
  }
  float inv = 1.f / S;
  for (int d = 0; d < 64; d++) {
    float o = 0.f;
    for (int s = 0; s < NSPLIT; s++)
      o += w[s] * Opart[((size_t)bh * NSPLIT + s) * 4096 + q * 64 + d];
    attn[(size_t)(b * 64 + q) * INNER + h * 64 + d] = f2bf(o * inv);
  }
}

// ---------------- host orchestration ----------------
extern "C" void kernel_launch(void* const* d_in, const int* in_sizes, int n_in,
                              void* d_out, int out_size, void* d_ws, size_t ws_size,
                              hipStream_t stream)
{
  (void)in_sizes; (void)n_in; (void)out_size; (void)ws_size;
  const int L = 6;
  const float* x        = (const float*)d_in[0];
  const float* latents  = (const float*)d_in[1];
  const float* frame_e  = (const float*)d_in[2];
  const float* time_e   = (const float*)d_in[3];
  const float* ln_m_g   = (const float*)d_in[4];
  const float* ln_m_b   = (const float*)d_in[5];
  const float* ln_l_g   = (const float*)d_in[6];
  const float* ln_l_b   = (const float*)d_in[7];
  const float* Wq       = (const float*)d_in[8];
  const float* Wk       = (const float*)d_in[9];
  const float* Wv       = (const float*)d_in[10];
  const float* Wo       = (const float*)d_in[11];
  const float* ff_g     = (const float*)d_in[12];
  const float* ff_b     = (const float*)d_in[13];
  const float* W1       = (const float*)d_in[14];
  const float* W2       = (const float*)d_in[15];
  const float* out_g    = (const float*)d_in[16];
  const float* out_b    = (const float*)d_in[17];

  char* ws = (char*)d_ws;
  auto carve = [&](size_t bytes) -> char* {
    char* p = ws; ws += (bytes + 255) & ~(size_t)255; return p;
  };
  unsigned short* xhat  = (unsigned short*)carve((size_t)NBATCH * NXTOK * DIMD * 2);
  unsigned short* Kbuf  = (unsigned short*)carve((size_t)NBATCH * NFTOK * INNER * 2);
  unsigned short* Vbuf  = (unsigned short*)carve((size_t)NBATCH * NFTOK * INNER * 2);
  unsigned short* WtK   = (unsigned short*)carve((size_t)INNER * DIMD * 2);
  unsigned short* WtV   = (unsigned short*)carve((size_t)INNER * DIMD * 2);
  float*          biasK = (float*)carve((size_t)INNER * 4);
  float*          biasV = (float*)carve((size_t)INNER * 4);
  unsigned short* lnb   = (unsigned short*)carve((size_t)256 * DIMD * 2);
  unsigned short* qb    = (unsigned short*)carve((size_t)256 * INNER * 2);
  unsigned short* attnb = (unsigned short*)carve((size_t)256 * INNER * 2);
  unsigned short* hln   = (unsigned short*)carve((size_t)256 * DIMD * 2);
  unsigned short* ffa   = (unsigned short*)carve((size_t)256 * FFH * 2);
  float* lat   = (float*)carve((size_t)256 * DIMD * 4);
  float* Opart = (float*)carve((size_t)32 * NSPLIT * 64 * 64 * 4);
  float* Mpart = (float*)carve((size_t)32 * NSPLIT * 64 * 4);
  float* Spart = (float*)carve((size_t)32 * NSPLIT * 64 * 4);

  const int NOREMAP = 1 << 30;
  const int TPS = (NTILES + NSPLIT - 1) / NSPLIT;   // 50

  prep_xhat_kernel<<<NBATCH * NXTOK, 256, 0, stream>>>(x, frame_e, time_e, xhat);
  init_lat_kernel<<<(NBATCH * NZ * DIMD) / 256, 256, 0, stream>>>(latents, lat);

  for (int i = 0; i < L; i++) {
    const float* Wq_i = Wq + (size_t)i * DIMD * INNER;
    const float* Wk_i = Wk + (size_t)i * DIMD * INNER;
    const float* Wv_i = Wv + (size_t)i * DIMD * INNER;
    const float* Wo_i = Wo + (size_t)i * INNER * DIMD;
    const float* W1_i = W1 + (size_t)i * DIMD * FFH;
    const float* W2_i = W2 + (size_t)i * FFH * DIMD;

    // fold media-LN gamma/beta into bf16, pre-transposed K/V weights + bias rows
    fold_weights_kernel<<<INNER, 256, 0, stream>>>(Wk_i, ln_m_g + i * DIMD, ln_m_b + i * DIMD, WtK, biasK);
    fold_weights_kernel<<<INNER, 256, 0, stream>>>(Wv_i, ln_m_g + i * DIMD, ln_m_b + i * DIMD, WtV, biasV);

    // latent layernorm (g_l, b_l baked in) -> bf16
    layernorm_rows_kernel<true><<<256, 256, 0, stream>>>(lat, ln_l_g + i * DIMD, ln_l_b + i * DIMD, lnb);

    dim3 gQ(2, INNER / 128);   // M=256, N=512
    gemm_bf16_kernel<0><<<gQ, 256, 0, stream>>>(lnb, DIMD, Wq_i, INNER,
                                                qb, INNER, DIMD, NOREMAP, 0, 0);
    gemm_bf16_kernel<0><<<gQ, 256, 0, stream>>>(lnb, DIMD, Wk_i, INNER,
                                                Kbuf, INNER, DIMD, NZ, NFTOK, NXTOK);
    gemm_bf16_kernel<0><<<gQ, 256, 0, stream>>>(lnb, DIMD, Wv_i, INNER,
                                                Vbuf, INNER, DIMD, NZ, NFTOK, NXTOK);

    // big K/V projections over media tokens (async double-buffered staging)
    dim3 gKV((NBATCH * NXTOK) / 128, INNER / 128);   // 784 x 4
    gemm_kv_kernel<<<gKV, 256, 0, stream>>>(xhat, WtK, biasK, Kbuf);
    gemm_kv_kernel<<<gKV, 256, 0, stream>>>(xhat, WtV, biasV, Vbuf);

    flash_attn_kernel<<<dim3(32, NSPLIT), 128, 0, stream>>>(qb, Kbuf, Vbuf, Opart, Mpart, Spart, TPS);
    attn_combine_kernel<<<32, 64, 0, stream>>>(Opart, Mpart, Spart, attnb);

    dim3 gO(2, DIMD / 128);    // M=256, N=768
    gemm_bf16_kernel<1><<<gO, 256, 0, stream>>>(attnb, INNER, Wo_i, DIMD,
                                                lat, DIMD, INNER, NOREMAP, 0, 0);

    layernorm_rows_kernel<true><<<256, 256, 0, stream>>>(lat, ff_g + i * DIMD, ff_b + i * DIMD, hln);

    dim3 gF1(2, FFH / 128);    // M=256, N=3072, GELU epilogue
    gemm_bf16_kernel<2><<<gF1, 256, 0, stream>>>(hln, DIMD, W1_i, FFH,
                                                 ffa, FFH, DIMD, NOREMAP, 0, 0);
    gemm_bf16_kernel<1><<<gO, 256, 0, stream>>>(ffa, FFH, W2_i, DIMD,
                                                lat, DIMD, FFH, NOREMAP, 0, 0);
  }

  layernorm_rows_kernel<false><<<256, 256, 0, stream>>>(lat, out_g, out_b, d_out);
}
